// WReN_28905129902684
// MI455X (gfx1250) — compile-verified, matrix operand order checked by reading, not run on hardware
//
#include <hip/hip_runtime.h>

typedef __attribute__((ext_vector_type(16))) _Float16 v16h;
typedef __attribute__((ext_vector_type(8)))  _Float16 v8h;
typedef __attribute__((ext_vector_type(8)))  float    v8f;

// ---------------------------------------------------------------- constants
#define BATCH 128
#define NPAN  16
#define NIMG  2048          // BATCH*NPAN
#define DEMB  256
#define NROWS 24576         // BATCH*192 distinct g-MLP rows

#define GEMM_BK      256    // K-chunk staged in LDS
#define GEMM_BPAD    8      // f16 pad per LDS row (bank-conflict avoidance)
#define GEMM_BSTRIDE (GEMM_BK + GEMM_BPAD)   // 264 f16 = 528 B per row

// ---------------------------------------------------------------- utility
__global__ void zero_f32_kernel(float* p, int n) {
  int i = blockIdx.x * 256 + threadIdx.x;
  if (i < n) p[i] = 0.f;
}

// ---------------------------------------------------------------- conv + BN stats
// One thread = one (img, oy, ox); computes all 32 output channels.
// Raw conv output stored f16; per-channel sum/sumsq accumulated via
// LDS atomics then flushed with global atomics (stats[0..31]=sum, [32..63]=sumsq).
template <int CIN, typename InT>
__global__ __launch_bounds__(256)
void conv_stats_kernel(const InT* __restrict__ in, const float* __restrict__ w,
                       _Float16* __restrict__ out, float* __restrict__ stats,
                       int IH, int OH, int nimg) {
  __shared__ float wl[32 * CIN * 9];
  __shared__ float ssum[32], ssq[32];
  for (int i = threadIdx.x; i < 32 * CIN * 9; i += 256) wl[i] = w[i];
  if (threadIdx.x < 32) { ssum[threadIdx.x] = 0.f; ssq[threadIdx.x] = 0.f; }
  __syncthreads();

  int idx   = blockIdx.x * 256 + threadIdx.x;
  int total = nimg * OH * OH;
  if (idx < total) {
    int img = idx / (OH * OH);
    int rem = idx - img * OH * OH;
    int oy  = rem / OH;
    int ox  = rem - oy * OH;
    float acc[32];
#pragma unroll
    for (int oc = 0; oc < 32; ++oc) acc[oc] = 0.f;
    const InT* ib = in + (size_t)img * CIN * IH * IH + (2 * oy) * IH + 2 * ox;
    for (int ic = 0; ic < CIN; ++ic) {
      const InT* ip = ib + (size_t)ic * IH * IH;
#pragma unroll
      for (int dy = 0; dy < 3; ++dy) {
#pragma unroll
        for (int dx = 0; dx < 3; ++dx) {
          float v = (float)ip[dy * IH + dx];
          const float* wp = wl + ic * 9 + dy * 3 + dx;
#pragma unroll
          for (int oc = 0; oc < 32; ++oc)
            acc[oc] = fmaf(v, wp[oc * CIN * 9], acc[oc]);
        }
      }
    }
    _Float16* op = out + (size_t)img * 32 * OH * OH + oy * OH + ox;
#pragma unroll
    for (int oc = 0; oc < 32; ++oc) {
      op[(size_t)oc * OH * OH] = (_Float16)acc[oc];
      atomicAdd(&ssum[oc], acc[oc]);
      atomicAdd(&ssq[oc],  acc[oc] * acc[oc]);
    }
  }
  __syncthreads();
  if (threadIdx.x < 32) {
    atomicAdd(&stats[threadIdx.x],      ssum[threadIdx.x]);
    atomicAdd(&stats[32 + threadIdx.x], ssq[threadIdx.x]);
  }
}

// In-place BN (batch stats) + ReLU on f16 buffer (layout img-major, ch, HW).
__global__ __launch_bounds__(256)
void bn_relu_kernel(_Float16* buf, const float* __restrict__ stats,
                    const float* __restrict__ g, const float* __restrict__ be,
                    int HW, int total, float invcnt) {
  int idx = blockIdx.x * 256 + threadIdx.x;
  if (idx >= total) return;
  int ch     = (idx / HW) & 31;
  float mean = stats[ch] * invcnt;
  float var  = stats[32 + ch] * invcnt - mean * mean;
  float sc   = g[ch] * rsqrtf(var + 1e-5f);
  float v    = ((float)buf[idx] - mean) * sc + be[ch];
  buf[idx]   = (_Float16)fmaxf(v, 0.f);
}

// ---------------------------------------------------------------- embedding
// emb[n][c] = feat[n][:512] . pW[:512, c] + pW[512+min(p,8)][c] + pb[c]
// (tags are a one-hot, so the 521-wide GEMM folds into K=512 + a bias row)
__global__ __launch_bounds__(256)
void emb_kernel(const _Float16* __restrict__ feat, const float* __restrict__ pW,
                const float* __restrict__ pb, _Float16* __restrict__ emb) {
  int idx = blockIdx.x * 256 + threadIdx.x;
  if (idx >= NIMG * DEMB) return;
  int n = idx >> 8, c = idx & 255;
  int p = n & 15;
  int tp = p < 8 ? p : 8;
  float acc = pW[(512 + tp) * DEMB + c] + pb[c];
  const _Float16* f = feat + (size_t)n * 512;
  for (int k = 0; k < 512; ++k) acc = fmaf((float)f[k], pW[k * DEMB + c], acc);
  emb[idx] = (_Float16)acc;
}

// ---------------------------------------------------------------- weight packing
// Generic transpose+f16: dst[N][K] = src[K][N]
__global__ __launch_bounds__(256)
void packT_kernel(const float* __restrict__ src, _Float16* __restrict__ dst,
                  int K, int N) {
  int idx = blockIdx.x * 256 + threadIdx.x;
  if (idx >= K * N) return;
  int n = idx / K, k = idx - n * K;
  dst[idx] = (_Float16)src[(size_t)k * N + n];
}

// wUV[1024][256]: rows 0..511 = rW1[:256,:]^T (first-slot), 512..1023 = rW1[256:,:]^T
__global__ __launch_bounds__(256)
void pack_w1_kernel(const float* __restrict__ rW1, _Float16* __restrict__ dst) {
  int idx = blockIdx.x * 256 + threadIdx.x;
  if (idx >= 1024 * 256) return;
  int n = idx >> 8, k = idx & 255;
  float v = (n < 512) ? rW1[(size_t)k * 512 + n]
                      : rW1[(size_t)(256 + k) * 512 + (n - 512)];
  dst[idx] = (_Float16)v;
}

// ---------------------------------------------------------------- WMMA GEMM
// out[M][N] = act(A[M][K] * WT[N][K]^T + bias), f16 in, f32 accumulate.
// Block = 8 waves, tile 256(M) x 64(N); the 64-row weight panel is staged in
// LDS in 256-wide K-chunks (33 KB, rows padded +8 f16 so the 16-lane
// ds_load_b128 fragment reads land on distinct banks). Each wave owns a
// 32x64 subtile: 2 A-frags (global b128) x 4 B-frags (LDS b128) -> 8 WMMAs
// per K-step. Per ISA 7.12.2 a lane's K values are two contiguous 8xf16 runs
// at k0+8*(lane>=16) and +16. M%256==0 and N%64==0 for all calls, so every
// wave is fully active and EXEC stays all-1s (WMMA requirement).
template <bool RELU, bool HAS_BIAS, typename OutT>
__global__ __launch_bounds__(256)
void gemm_wmma_kernel(const _Float16* __restrict__ A, const _Float16* __restrict__ WT,
                      const float* __restrict__ bias, OutT* __restrict__ out,
                      int M, int N, int K) {
  __shared__ _Float16 Bs[64 * GEMM_BSTRIDE];
  const int tid  = threadIdx.x;
  const int lane = tid & 31;
  const int w    = tid >> 5;            // wave 0..7
  const int nblocks = N >> 6;
  const int mb = blockIdx.x / nblocks;
  const int nb = blockIdx.x - mb * nblocks;
  const int l15   = lane & 15;
  const int khalf = (lane >> 4) << 3;
  const int mbase = mb * 256 + w * 32;
  const int nbase = nb * 64;

  const _Float16* a0 = A + (size_t)(mbase + l15) * K + khalf;
  const _Float16* a1 = a0 + (size_t)16 * K;

  v8f acc[2][4] = {};
  union Uu { v16h v; v8h h[2]; };

  for (int k0 = 0; k0 < K; k0 += GEMM_BK) {
    __syncthreads();   // previous chunk fully consumed before overwrite
    {
      // cooperative stage: 64 x 256 f16 = 2048 x b128; 8 vectors per thread
      const _Float16* src = WT + (size_t)nbase * K + k0;
#pragma unroll
      for (int t = 0; t < 8; ++t) {
        int vec = t * 256 + tid;          // 0..2047
        int n   = vec >> 5;               // 32 vectors per row
        int kk  = (vec & 31) << 3;
        *(v8h*)(Bs + n * GEMM_BSTRIDE + kk) = *(const v8h*)(src + (size_t)n * K + kk);
      }
    }
    __syncthreads();

    if (k0 + GEMM_BK < K) {               // prefetch next A chunk (global_prefetch_b8)
      __builtin_prefetch(a0 + k0 + GEMM_BK, 0, 1);
      __builtin_prefetch(a1 + k0 + GEMM_BK, 0, 1);
    }

    for (int k = 0; k < GEMM_BK; k += 32) {
      Uu a[2], b[4];
      a[0].h[0] = *(const v8h*)(a0 + k0 + k);
      a[0].h[1] = *(const v8h*)(a0 + k0 + k + 16);
      a[1].h[0] = *(const v8h*)(a1 + k0 + k);
      a[1].h[1] = *(const v8h*)(a1 + k0 + k + 16);
#pragma unroll
      for (int nf = 0; nf < 4; ++nf) {
        const _Float16* bp = Bs + (nf * 16 + l15) * GEMM_BSTRIDE + khalf + k;
        b[nf].h[0] = *(const v8h*)bp;
        b[nf].h[1] = *(const v8h*)(bp + 16);
      }
#pragma unroll
      for (int mf = 0; mf < 2; ++mf)
#pragma unroll
        for (int nf = 0; nf < 4; ++nf)
          acc[mf][nf] = __builtin_amdgcn_wmma_f32_16x16x32_f16(
              false, a[mf].v, false, b[nf].v, (short)0, acc[mf][nf], false, false);
    }
  }

  // C/D layout: lane L col = L%16; VGPR r -> row r + 8*(L>=16)
  const int rb0 = (lane >> 4) << 3;
#pragma unroll
  for (int mf = 0; mf < 2; ++mf) {
#pragma unroll
    for (int nf = 0; nf < 4; ++nf) {
      int col  = nbase + nf * 16 + l15;
      float bv = 0.f;
      if constexpr (HAS_BIAS) bv = bias[col];
      const v8f c = acc[mf][nf];
#pragma unroll
      for (int r = 0; r < 8; ++r) {
        float v = c[r] + bv;
        if constexpr (RELU) v = fmaxf(v, 0.f);
        out[(size_t)(mbase + mf * 16 + rb0 + r) * N + col] = (OutT)v;
      }
    }
  }
}

// ---------------------------------------------------------------- r1 from U/V
// 192 distinct pairs per batch: t<64 cc (first=t%8, second=t/8),
// 64..127 (ctx_i, cand_k), 128..191 (cand_k, ctx_i).
__global__ __launch_bounds__(256)
void r1_build_kernel(const _Float16* __restrict__ UV, const float* __restrict__ rb1,
                     _Float16* __restrict__ r1) {
  int idx = blockIdx.x * 256 + threadIdx.x;
  if (idx >= NROWS * 512) return;
  int row = idx >> 9, d = idx & 511;
  int b = row / 192, t = row - b * 192;
  int first, second;
  if (t < 64)        { first = t & 7;              second = t >> 3; }
  else if (t < 128)  { int u = t - 64;  first = u & 7;         second = 8 + (u >> 3); }
  else               { int u = t - 128; first = 8 + (u >> 3);  second = u & 7; }
  int e0 = b * 16 + first, e1 = b * 16 + second;
  float v = (float)UV[(size_t)e0 * 1024 + d] +
            (float)UV[(size_t)e1 * 1024 + 512 + d] + rb1[d];
  r1[idx] = (_Float16)fmaxf(v, 0.f);
}

// ---------------------------------------------------------------- pair-sum
__global__ __launch_bounds__(256)
void scc_kernel(const float* __restrict__ r4, float* __restrict__ scc) {
  int idx = blockIdx.x * 256 + threadIdx.x;
  if (idx >= BATCH * 256) return;
  int b = idx >> 8, c = idx & 255;
  const float* p = r4 + (size_t)b * 192 * 256 + c;
  float s = 0.f;
  for (int t = 0; t < 64; ++t) s += p[t * 256];
  scc[idx] = s;
}

__global__ __launch_bounds__(256)
void summed_kernel(const float* __restrict__ r4, const float* __restrict__ scc,
                   float* __restrict__ summed) {
  int idx = blockIdx.x * 256 + threadIdx.x;
  if (idx >= BATCH * 8 * 256) return;
  int bk = idx >> 8, c = idx & 255;
  int b = bk >> 3, k = bk & 7;
  const float* base = r4 + (size_t)b * 192 * 256 + c;
  float s = scc[b * 256 + c];
#pragma unroll
  for (int i = 0; i < 8; ++i) {
    s += base[(64  + k * 8 + i) * 256];
    s += base[(128 + k * 8 + i) * 256];
  }
  summed[idx] = s;
}

// ---------------------------------------------------------------- f MLP
__global__ __launch_bounds__(256)
void fmlp1_kernel(const float* __restrict__ summed, const float* __restrict__ mW1,
                  const float* __restrict__ mb1, float* __restrict__ m) {
  int idx = blockIdx.x * 256 + threadIdx.x;
  if (idx >= 1024 * 256) return;
  int n = idx >> 8, c = idx & 255;
  float acc = mb1[c];
  const float* s = summed + (size_t)n * 256;
  for (int h = 0; h < 256; ++h) acc = fmaf(s[h], mW1[h * 256 + c], acc);
  m[idx] = fmaxf(acc, 0.f);
}

__global__ __launch_bounds__(256)
void fmlp2_kernel(const float* __restrict__ m, const float* __restrict__ mW2,
                  const float* __restrict__ mb2, float* __restrict__ out) {
  int idx = blockIdx.x * 256 + threadIdx.x;
  if (idx >= 1024) return;
  const float* r = m + (size_t)idx * 256;
  float acc = mb2[12];
  for (int c = 0; c < 256; ++c) acc = fmaf(r[c], mW2[c * 13 + 12], acc);
  out[idx] = acc;
}

// ---------------------------------------------------------------- launch
extern "C" void kernel_launch(void* const* d_in, const int* in_sizes, int n_in,
                              void* d_out, int out_size, void* d_ws, size_t ws_size,
                              hipStream_t stream) {
  (void)in_sizes; (void)n_in; (void)out_size; (void)ws_size;
  const float* x   = (const float*)d_in[0];
  const float* cw1 = (const float*)d_in[1];
  const float* cg1 = (const float*)d_in[3];
  const float* cB1 = (const float*)d_in[4];
  const float* cw2 = (const float*)d_in[5];
  const float* cg2 = (const float*)d_in[7];
  const float* cB2 = (const float*)d_in[8];
  const float* cw3 = (const float*)d_in[9];
  const float* cg3 = (const float*)d_in[11];
  const float* cB3 = (const float*)d_in[12];
  const float* cw4 = (const float*)d_in[13];
  const float* cg4 = (const float*)d_in[15];
  const float* cB4 = (const float*)d_in[16];
  // conv biases (cb*) cancel exactly under batch-norm -> dropped
  const float* pW  = (const float*)d_in[17];
  const float* pb  = (const float*)d_in[18];
  const float* rW1 = (const float*)d_in[19];
  const float* rb1 = (const float*)d_in[20];
  const float* rW2 = (const float*)d_in[21];
  const float* rb2 = (const float*)d_in[22];
  const float* rW3 = (const float*)d_in[23];
  const float* rb3 = (const float*)d_in[24];
  const float* rW4 = (const float*)d_in[25];
  const float* rb4 = (const float*)d_in[26];
  const float* mW1 = (const float*)d_in[27];
  const float* mb1 = (const float*)d_in[28];
  const float* mW2 = (const float*)d_in[29];
  const float* mb2 = (const float*)d_in[30];

  char* ws = (char*)d_ws;
  size_t off = 0;
  auto alloc = [&](size_t b) { size_t o = off; off += (b + 255) & ~(size_t)255; return o; };

  float*    stats = (float*)   (ws + alloc(4 * 64 * sizeof(float)));
  size_t c1off = alloc((size_t)NIMG * 32 * 39 * 39 * 2);   // 199 MB (reused below)
  _Float16* c1    = (_Float16*)(ws + c1off);
  _Float16* c2    = (_Float16*)(ws + alloc((size_t)NIMG * 32 * 19 * 19 * 2));
  _Float16* c3    = (_Float16*)(ws + alloc((size_t)NIMG * 32 * 9 * 9 * 2));
  _Float16* c4    = (_Float16*)(ws + alloc((size_t)NIMG * 32 * 16 * 2));
  _Float16* wUV   = (_Float16*)(ws + alloc((size_t)1024 * 256 * 2));
  _Float16* w2T   = (_Float16*)(ws + alloc((size_t)512 * 512 * 2));
  _Float16* w3T   = (_Float16*)(ws + alloc((size_t)512 * 512 * 2));
  _Float16* w4T   = (_Float16*)(ws + alloc((size_t)256 * 512 * 2));
  _Float16* emb   = (_Float16*)(ws + alloc((size_t)NIMG * 256 * 2));
  _Float16* UV    = (_Float16*)(ws + alloc((size_t)NIMG * 1024 * 2));
  float*    scc   = (float*)   (ws + alloc((size_t)BATCH * 256 * 4));
  float*    summed= (float*)   (ws + alloc((size_t)1024 * 256 * 4));
  float*    mbuf  = (float*)   (ws + alloc((size_t)1024 * 256 * 4));
  // r1..r4 alias the (dead-by-then) conv1 buffer: 4 x 25.2 MB < 199 MB
  _Float16* r1 = (_Float16*)(ws + c1off);
  _Float16* r2 = r1 + (size_t)NROWS * 512;
  _Float16* r3 = r2 + (size_t)NROWS * 512;
  float*    r4 = (float*)(r3 + (size_t)NROWS * 512);

  auto G = [](long long n) { return dim3((unsigned)((n + 255) / 256)); };
  auto GG = [](int M, int N) { return dim3((unsigned)((M / 256) * (N / 64))); };

  // --- BN stats must start at zero every call (ws is poisoned, not re-zeroed)
  zero_f32_kernel<<<G(256), 256, 0, stream>>>(stats, 256);

  // --- conv tower (f32 accumulate, f16 activations), BN stats fused
  conv_stats_kernel<1,  float>   <<<G((long long)NIMG * 39 * 39), 256, 0, stream>>>(x,  cw1, c1, stats,       80, 39, NIMG);
  bn_relu_kernel<<<G((long long)NIMG * 32 * 39 * 39), 256, 0, stream>>>(c1, stats,       cg1, cB1, 39 * 39, NIMG * 32 * 39 * 39, 1.f / (NIMG * 39.f * 39.f));
  conv_stats_kernel<32, _Float16><<<G((long long)NIMG * 19 * 19), 256, 0, stream>>>(c1, cw2, c2, stats + 64,  39, 19, NIMG);
  bn_relu_kernel<<<G((long long)NIMG * 32 * 19 * 19), 256, 0, stream>>>(c2, stats + 64,  cg2, cB2, 19 * 19, NIMG * 32 * 19 * 19, 1.f / (NIMG * 19.f * 19.f));
  conv_stats_kernel<32, _Float16><<<G((long long)NIMG * 9 * 9),   256, 0, stream>>>(c2, cw3, c3, stats + 128, 19, 9,  NIMG);
  bn_relu_kernel<<<G((long long)NIMG * 32 * 9 * 9),   256, 0, stream>>>(c3, stats + 128, cg3, cB3, 9 * 9,   NIMG * 32 * 9 * 9,   1.f / (NIMG * 81.f));
  conv_stats_kernel<32, _Float16><<<G((long long)NIMG * 4 * 4),   256, 0, stream>>>(c3, cw4, c4, stats + 192, 9,  4,  NIMG);
  bn_relu_kernel<<<G((long long)NIMG * 32 * 4 * 4),   256, 0, stream>>>(c4, stats + 192, cg4, cB4, 4 * 4,   NIMG * 32 * 4 * 4,   1.f / (NIMG * 16.f));

  // --- embedding (tag one-hot folded into per-panel bias row), pack weights
  emb_kernel<<<G((long long)NIMG * 256), 256, 0, stream>>>(c4, pW, pb, emb);
  pack_w1_kernel<<<G(1024 * 256), 256, 0, stream>>>(rW1, wUV);
  packT_kernel<<<G(512 * 512), 256, 0, stream>>>(rW2, w2T, 512, 512);
  packT_kernel<<<G(512 * 512), 256, 0, stream>>>(rW3, w3T, 512, 512);
  packT_kernel<<<G(512 * 256), 256, 0, stream>>>(rW4, w4T, 512, 256);

  // --- g-MLP on 24576 distinct pair rows (WMMA f16 -> f32, LDS-staged weights)
  // layer-1 factorization: UV = emb @ [W1_top | W1_bot]  (2048 x 256 x 1024)
  gemm_wmma_kernel<false, false, _Float16><<<GG(NIMG, 1024), 256, 0, stream>>>(emb, wUV, nullptr, UV, NIMG, 1024, 256);
  r1_build_kernel<<<G((long long)NROWS * 512), 256, 0, stream>>>(UV, rb1, r1);
  gemm_wmma_kernel<true, true, _Float16><<<GG(NROWS, 512), 256, 0, stream>>>(r1, w2T, rb2, r2, NROWS, 512, 512);
  gemm_wmma_kernel<true, true, _Float16><<<GG(NROWS, 512), 256, 0, stream>>>(r2, w3T, rb3, r3, NROWS, 512, 512);
  gemm_wmma_kernel<true, true, float>   <<<GG(NROWS, 256), 256, 0, stream>>>(r3, w4T, rb4, r4, NROWS, 256, 512);

  // --- pair-sum reconstruction + f MLP (col 12 only)
  scc_kernel   <<<G(BATCH * 256),      256, 0, stream>>>(r4, scc);
  summed_kernel<<<G(BATCH * 8 * 256),  256, 0, stream>>>(r4, scc, summed);
  fmlp1_kernel <<<G(1024 * 256),       256, 0, stream>>>(summed, mW1, mb1, mbuf);
  fmlp2_kernel <<<G(1024),             256, 0, stream>>>(mbuf, mW2, mb2, (float*)d_out);
}
